// StreamHeightSequence_19782619365381
// MI455X (gfx1250) — compile-verified, matrix operand order checked by reading, not run on hardware
//
#include <hip/hip_runtime.h>
#include <cstdint>
#include <cstddef>

// ---------------------------------------------------------------------------
// Seq2seq LSTM (3-layer encoder + 3-layer autoregressive decoder) for MI455X.
//
// MI455X-specific design:
//  * bf16 WMMA (v_wmma_f32_16x16x32_bf16, fp32 accum) for all recurrent GEMMs;
//    fp32 cell state + gate nonlinearities.
//  * Packed weights ~84 MB bf16 -> resident in the 192 MB L2; every
//    sequential step reads weights at L2 bandwidth, not HBM (23.3 TB/s).
//  * Input GEMM fused into the recurrent GEMM (A = [x_t | h_prev]).
//  * Persistent kernels (64 WGs x 8 wave32) + device-scope grid barrier walk
//    the 512+511 step sequential chain without ~3000 kernel launches.
//  * NEW: per-WG weight slabs are staged into LDS by the Tensor Data Mover
//    (tensor_load_to_lds, TENSORcnt, double-buffered 2x32KB in the 320KB WGP
//    LDS), eliminating the 8x redundant per-wave B-fragment loads; waves read
//    B fragments via ds_load_b128.
//
// Workspace requirement: ~94 MB.
// ---------------------------------------------------------------------------

#define HDIM 1024
#define BDIM 128
#define TDIM 512
#define G4H  4096
#define NWG  64
#define NTHREADS 256
#define CKT  8                                   // k-tiles per LDS chunk
#define LDSWB_HALVES (2 * 4 * CKT * 512)         // double buffer, 4 gates

typedef __bf16 bf16x16 __attribute__((ext_vector_type(16)));
typedef float  floatx8 __attribute__((ext_vector_type(8)));
typedef unsigned int v4u __attribute__((ext_vector_type(4)));
typedef int          v8i __attribute__((ext_vector_type(8)));
typedef int          v4i __attribute__((ext_vector_type(4)));

#if defined(__has_builtin)
#  if __has_builtin(__builtin_amdgcn_tensor_load_to_lds) && \
      __has_builtin(__builtin_amdgcn_s_wait_tensorcnt)
#    define USE_TDM 1
#  endif
#endif
#ifndef USE_TDM
#  define USE_TDM 0
#endif

__device__ __forceinline__ unsigned short f2bf(float f) {
    unsigned u = __builtin_bit_cast(unsigned, f);
    u += 0x7FFFu + ((u >> 16) & 1u);             // round-to-nearest-even
    return (unsigned short)(u >> 16);
}
__device__ __forceinline__ float fsig(float x) { return 1.0f / (1.0f + __expf(-x)); }

// ---------------------------------------------------------------------------
// Device-scope grid barrier (monotone counter; persistent-kernel lockstep).
// ---------------------------------------------------------------------------
__device__ __forceinline__ void grid_barrier(unsigned* bar, unsigned target) {
    __threadfence();
    __syncthreads();
    if (threadIdx.x == 0) {
        __hip_atomic_fetch_add(bar, 1u, __ATOMIC_ACQ_REL, __HIP_MEMORY_SCOPE_AGENT);
        while (__hip_atomic_load(bar, __ATOMIC_ACQUIRE, __HIP_MEMORY_SCOPE_AGENT) < target) {
            __builtin_amdgcn_s_sleep(2);
        }
    }
    __syncthreads();
    __threadfence();
}

#if USE_TDM
// ---------------------------------------------------------------------------
// Issue one TDM 2D tile load: 4 gate rows x ckBytes each, row stride =
// gateStrideBytes, into contiguous LDS at ldsOff. D# per CDNA5 ISA.
// data_size = 8 bytes (code 3). Groups 2/3 zero (tensor <= 2D).
// ---------------------------------------------------------------------------
__device__ __forceinline__ void tdm_issue(const unsigned short* gbase,
                                          size_t gateStrideBytes,
                                          unsigned ldsOff, int ckBytes) {
    const unsigned long long ga = (unsigned long long)(uintptr_t)gbase;
    const unsigned td0 = (unsigned)(ckBytes >> 3);          // row len, 8B units
    const unsigned long long st = gateStrideBytes >> 3;     // row stride, 8B units
    v4u g0;
    g0.x = 1u;                                              // count=1, user D#
    g0.y = ldsOff;                                          // lds_addr
    g0.z = (unsigned)ga;                                    // global_addr[31:0]
    g0.w = (unsigned)((ga >> 32) & 0x01FFFFFFu) | (2u << 30); // addr[56:32]|type=2
    v8i g1;
    g1[0] = 0x30000;                          // wg_mask=0, data_size=3 (8B)
    g1[1] = (int)((td0 & 0xFFFFu) << 16);     // tensor_dim0[15:0] @ bits 63:48
    g1[2] = (int)((td0 >> 16) | (4u << 16));  // tensor_dim0[31:16], tensor_dim1=4
    g1[3] = (int)(td0 << 16);                 // tensor_dim1 hi=0, tile_dim0=td0
    g1[4] = 4;                                // tile_dim1=4 gates, tile_dim2=0
    g1[5] = (int)(st & 0xFFFFFFFFu);          // tensor_dim0_stride[31:0]
    g1[6] = (int)((st >> 32) & 0xFFFFu);      // stride[47:32], dim1_stride lo=0
    g1[7] = 0;
    v4i z4 = {0, 0, 0, 0};
#if defined(__clang_major__) && (__clang_major__ >= 23)
    v8i z8 = {0, 0, 0, 0, 0, 0, 0, 0};
    __builtin_amdgcn_tensor_load_to_lds(g0, g1, z4, z4, z8, 0);
#else
    __builtin_amdgcn_tensor_load_to_lds(g0, g1, z4, z4, 0);
#endif
}
#endif // USE_TDM

// ---------------------------------------------------------------------------
// One fused LSTM layer-step:
//   gates[128,4096] = [A0 | A1] x Bpacked  (+ bias) (+ optional rank-1 x-in)
//   then cell nonlinearity, c-state update, bf16 repack of h.
// A fragments: [ktile][mblock(8)][lane(32)][16 halves]; B (weights):
// [ntile(256)][ktile][lane(32)][16 halves] -- wave32 WMMA 16x16x32 layout.
// WG wg owns hidden cols wg*16..+15 (gate tiles at ntile g*64+wg); wave w owns
// batch rows 16w..16w+15. B slabs staged into LDS by the TDM, double-buffered.
// ---------------------------------------------------------------------------
__device__ __forceinline__ void layer_step(
    int wg, int wave, int lane,
    const unsigned short* __restrict__ a0, int nk0,
    const unsigned short* __restrict__ a1, int nk1,
    const unsigned short* __restrict__ wB,
    unsigned short* lds_wb,                   // [LDSWB_HALVES] shared staging
    const float* __restrict__ bias,           // [4096] = bih+bhh
    const float* __restrict__ xin,            // optional [128] scalar input
    const float* __restrict__ wih_col,        // optional [4096] rank-1 weight
    float* __restrict__ c_state,              // [128][1024] fp32
    unsigned short* __restrict__ h_pack_out,  // packed bf16 A-fragments
    float* __restrict__ h_plain_out)          // optional [128][1024] fp32
{
    floatx8 acc0 = {}, acc1 = {}, acc2 = {}, acc3 = {};
    const int nkTot = nk0 + nk1;

#if USE_TDM
    const size_t gateStrideBytes = (size_t)64 * nkTot * 512 * 2;
    const unsigned ldsBase = (unsigned)(uintptr_t)lds_wb;
    {   // prologue: stage chunk 0 into buffer 0
        const int ck0 = nkTot < CKT ? nkTot : CKT;
        if (wave == 0) {
            tdm_issue(wB + (size_t)wg * nkTot * 512, gateStrideBytes,
                      ldsBase, ck0 * 1024);
            __builtin_amdgcn_s_wait_tensorcnt(0);
        }
    }
    __syncthreads();
    int buf = 0;
    for (int c0 = 0; c0 < nkTot; c0 += CKT) {
        const int ck = (nkTot - c0) < CKT ? (nkTot - c0) : CKT;
        const int cn = c0 + ck;
        if (cn < nkTot && wave == 0) {        // async-stage next chunk
            const int ckn = (nkTot - cn) < CKT ? (nkTot - cn) : CKT;
            tdm_issue(wB + ((size_t)wg * nkTot + cn) * 512, gateStrideBytes,
                      ldsBase + (unsigned)((buf ^ 1) * (4 * CKT * 512 * 2)),
                      ckn * 1024);
        }
        const unsigned short* lb = lds_wb + (size_t)buf * (4 * CKT * 512);
        for (int j = 0; j < ck; ++j) {
            const int kt = c0 + j;
            const unsigned short* ab = (kt < nk0)
                ? a0 + (((size_t)kt * 8 + wave) * 32 + lane) * 16
                : a1 + (((size_t)(kt - nk0) * 8 + wave) * 32 + lane) * 16;
            bf16x16 a = *(const bf16x16*)ab;
            const unsigned short* bb = lb + (size_t)j * 512 + (size_t)lane * 16;
            bf16x16 b0 = *(const bf16x16*)(bb);
            bf16x16 b1 = *(const bf16x16*)(bb + (size_t)ck * 512);
            bf16x16 b2 = *(const bf16x16*)(bb + (size_t)2 * ck * 512);
            bf16x16 b3 = *(const bf16x16*)(bb + (size_t)3 * ck * 512);
            acc0 = __builtin_amdgcn_wmma_f32_16x16x32_bf16(false, a, false, b0, (short)0, acc0, false, false);
            acc1 = __builtin_amdgcn_wmma_f32_16x16x32_bf16(false, a, false, b1, (short)0, acc1, false, false);
            acc2 = __builtin_amdgcn_wmma_f32_16x16x32_bf16(false, a, false, b2, (short)0, acc2, false, false);
            acc3 = __builtin_amdgcn_wmma_f32_16x16x32_bf16(false, a, false, b3, (short)0, acc3, false, false);
        }
        if (cn < nkTot && wave == 0)
            __builtin_amdgcn_s_wait_tensorcnt(0);
        __syncthreads();
        buf ^= 1;
    }
#else
    // Fallback: direct global-load B fragments (L2-resident weights).
    const size_t gstride = (size_t)64 * nkTot * 512;
    for (int kt = 0; kt < nkTot; ++kt) {
        const unsigned short* ab = (kt < nk0)
            ? a0 + (((size_t)kt * 8 + wave) * 32 + lane) * 16
            : a1 + (((size_t)(kt - nk0) * 8 + wave) * 32 + lane) * 16;
        bf16x16 a = *(const bf16x16*)ab;
        const unsigned short* bb =
            wB + ((size_t)wg * nkTot + kt) * 512 + (size_t)lane * 16;
        bf16x16 b0 = *(const bf16x16*)(bb);
        bf16x16 b1 = *(const bf16x16*)(bb + gstride);
        bf16x16 b2 = *(const bf16x16*)(bb + 2 * gstride);
        bf16x16 b3 = *(const bf16x16*)(bb + 3 * gstride);
        acc0 = __builtin_amdgcn_wmma_f32_16x16x32_bf16(false, a, false, b0, (short)0, acc0, false, false);
        acc1 = __builtin_amdgcn_wmma_f32_16x16x32_bf16(false, a, false, b1, (short)0, acc1, false, false);
        acc2 = __builtin_amdgcn_wmma_f32_16x16x32_bf16(false, a, false, b2, (short)0, acc2, false, false);
        acc3 = __builtin_amdgcn_wmma_f32_16x16x32_bf16(false, a, false, b3, (short)0, acc3, false, false);
    }
    (void)lds_wb;
#endif

    // ----- fused LSTM cell epilogue -----
    const int n     = wg * 16 + (lane & 15);       // hidden column
    const int rowHi = (lane >> 4) << 3;            // C layout: lanes 16-31 -> M+8
    const float bi = bias[n],          bf_ = bias[1024 + n];
    const float bg = bias[2048 + n],   bo  = bias[3072 + n];

#pragma unroll
    for (int r = 0; r < 8; ++r) {
        const int b = wave * 16 + r + rowHi;       // batch row
        float gi = acc0[r] + bi;
        float gf = acc1[r] + bf_;
        float gg = acc2[r] + bg;
        float go = acc3[r] + bo;
        if (xin) {
            const float xv = xin[b];
            gi = fmaf(xv, wih_col[n],          gi);
            gf = fmaf(xv, wih_col[1024 + n],   gf);
            gg = fmaf(xv, wih_col[2048 + n],   gg);
            go = fmaf(xv, wih_col[3072 + n],   go);
        }
        const float ig = fsig(gi), fg = fsig(gf);
        const float gt = tanhf(gg), og = fsig(go);
        const size_t cidx = (size_t)b * HDIM + n;
        const float c = fmaf(fg, c_state[cidx], ig * gt);
        c_state[cidx] = c;
        const float h = og * tanhf(c);
        if (h_plain_out) h_plain_out[cidx] = h;
        // repack h into the bf16 A-fragment layout for the next GEMM
        const int kt2 = n >> 5, ki = n & 31;
        const int fl = (b & 15) | (((ki >> 3) & 1) << 4);
        const int fe = (ki & 7) | ((ki >> 4) << 3);
        h_pack_out[(((size_t)kt2 * 8 + (b >> 4)) * 32 + fl) * 16 + fe] = f2bf(h);
    }
}

// ---------------------------------------------------------------------------
// Packing kernels (run once per launch; weights stay L2-resident afterwards)
// ---------------------------------------------------------------------------
__global__ void pack_weights_kernel(const float* __restrict__ Wx, int Kx,
                                    const float* __restrict__ Wh, int Kh,
                                    unsigned short* __restrict__ out,
                                    int nkTot, int total) {
    int idx = blockIdx.x * blockDim.x + threadIdx.x;
    if (idx >= total) return;
    const int fe = idx & 15;
    const int ln = (idx >> 4) & 31;
    const int rem = idx >> 9;
    const int kt = rem % nkTot;
    const int nt = rem / nkTot;
    const int n  = nt * 16 + (ln & 15);
    const int ki = (fe & 7) | ((ln >> 4) << 3) | ((fe >> 3) << 4);
    const int k  = kt * 32 + ki;
    float v;
    if (k < Kx) v = Wx[(size_t)n * Kx + k];
    else        v = Wh[(size_t)n * Kh + (k - Kx)];
    out[idx] = f2bf(v);
}

__global__ void pack_bias_kernel(const float* __restrict__ a,
                                 const float* __restrict__ b,
                                 float* __restrict__ out) {
    int n = blockIdx.x * blockDim.x + threadIdx.x;
    if (n < G4H) out[n] = a[n] + b[n];
}

__global__ void pack_x_kernel(const float* __restrict__ x,
                              unsigned short* __restrict__ out) {
    int idx = blockIdx.x * blockDim.x + threadIdx.x;     // [t][mb][lane][fe]
    if (idx >= TDIM * 8 * 32 * 16) return;
    const int fe = idx & 15;
    const int ln = (idx >> 4) & 31;
    const int mb = (idx >> 9) & 7;
    const int t  = idx >> 12;
    const int b  = mb * 16 + (ln & 15);
    const int ki = (fe & 7) | ((ln >> 4) << 3) | ((fe >> 3) << 4);
    out[idx] = f2bf(x[((size_t)b * TDIM + t) * 32 + ki]);
}

// ---------------------------------------------------------------------------
// Persistent encoder: 512 timesteps x 3 fused layer-steps, grid barriers.
// ---------------------------------------------------------------------------
struct EncP {
    const unsigned short *wB0, *wB1, *wB2;
    const float *bias0, *bias1, *bias2;
    const unsigned short *xp;
    unsigned short *h0[2], *h1[2], *h2[2];
    float *c0, *c1, *c2;
    unsigned *bar;
};

__global__ void __launch_bounds__(NTHREADS) enc_kernel(EncP p) {
    __shared__ unsigned short lds_wb[LDSWB_HALVES];
    const int wg = blockIdx.x, wave = threadIdx.x >> 5, lane = threadIdx.x & 31;
    unsigned tgt = 0;
    for (int t = 0; t < TDIM; ++t) {
        const int rd = t & 1, wr = rd ^ 1;
        layer_step(wg, wave, lane, p.xp + (size_t)t * 8 * 32 * 16, 1, p.h0[rd], 32,
                   p.wB0, lds_wb, p.bias0, nullptr, nullptr, p.c0, p.h0[wr], nullptr);
        tgt += NWG; grid_barrier(p.bar, tgt);
        layer_step(wg, wave, lane, p.h0[wr], 32, p.h1[rd], 32,
                   p.wB1, lds_wb, p.bias1, nullptr, nullptr, p.c1, p.h1[wr], nullptr);
        tgt += NWG; grid_barrier(p.bar, tgt);
        layer_step(wg, wave, lane, p.h1[wr], 32, p.h2[rd], 32,
                   p.wB2, lds_wb, p.bias2, nullptr, nullptr, p.c2, p.h2[wr], nullptr);
        tgt += NWG; grid_barrier(p.bar, tgt);
    }
}

// ---------------------------------------------------------------------------
// Persistent decoder: 511 steps, 3 fused layers + projection + feedback.
// Starts from encoder final states (shared buffers, parity index 0).
// ---------------------------------------------------------------------------
struct DecP {
    const unsigned short *wB0, *wB1, *wB2;
    const float *bias0, *bias1, *bias2;
    const float *wih0col;            // dec_Wih0 [4096]
    const float *Wu, *bu;
    const float *y; const int *force;
    unsigned short *h0[2], *h1[2], *h2[2];
    float *c0, *c1, *c2;
    float *h_plain, *x_scalar;
    float *out;
    unsigned *bar;
};

__global__ void __launch_bounds__(NTHREADS) dec_kernel(DecP p) {
    __shared__ unsigned short lds_wb[LDSWB_HALVES];
    float* red = (float*)lds_wb;      // reduction scratch aliases staging LDS
    const int wg = blockIdx.x, wave = threadIdx.x >> 5, lane = threadIdx.x & 31;
    unsigned tgt = 0;

    if (wg == 0 && threadIdx.x < BDIM)                 // x0 = y[:,0]
        p.x_scalar[threadIdx.x] = p.y[(size_t)threadIdx.x * TDIM];
    tgt += NWG; grid_barrier(p.bar, tgt);

    for (int s = 0; s < TDIM - 1; ++s) {
        const int rd = s & 1, wr = rd ^ 1;
        layer_step(wg, wave, lane, nullptr, 0, p.h0[rd], 32,
                   p.wB0, lds_wb, p.bias0, p.x_scalar, p.wih0col, p.c0, p.h0[wr], nullptr);
        tgt += NWG; grid_barrier(p.bar, tgt);
        layer_step(wg, wave, lane, p.h0[wr], 32, p.h1[rd], 32,
                   p.wB1, lds_wb, p.bias1, nullptr, nullptr, p.c1, p.h1[wr], nullptr);
        tgt += NWG; grid_barrier(p.bar, tgt);
        layer_step(wg, wave, lane, p.h1[wr], 32, p.h2[rd], 32,
                   p.wB2, lds_wb, p.bias2, nullptr, nullptr, p.c2, p.h2[wr], p.h_plain);
        tgt += NWG; grid_barrier(p.bar, tgt);

        if (wg == 0) {                                  // out = h2 @ Wu.T + bu
            const int b = threadIdx.x >> 1, half = threadIdx.x & 1;
            const float* hv = p.h_plain + (size_t)b * HDIM + half * 512;
            const float* wv = p.Wu + half * 512;
            float acc = 0.f;
            for (int j = 0; j < 512; j += 4) {
                acc = fmaf(hv[j],     wv[j],     acc);
                acc = fmaf(hv[j + 1], wv[j + 1], acc);
                acc = fmaf(hv[j + 2], wv[j + 2], acc);
                acc = fmaf(hv[j + 3], wv[j + 3], acc);
            }
            red[threadIdx.x] = acc;
            __syncthreads();
            if (half == 0) {
                const float o = red[threadIdx.x] + red[threadIdx.x + 1] + p.bu[0];
                p.out[(size_t)b * TDIM + (s + 1)] = o;
                p.x_scalar[b] = (p.force[s] > 0) ? p.y[(size_t)b * TDIM + (s + 1)] : o;
            }
            __syncthreads();
        }
        tgt += NWG; grid_barrier(p.bar, tgt);
    }
}

// ---------------------------------------------------------------------------
extern "C" void kernel_launch(void* const* d_in, const int* in_sizes, int n_in,
                              void* d_out, int out_size, void* d_ws, size_t ws_size,
                              hipStream_t stream) {
    (void)in_sizes; (void)n_in; (void)ws_size;
    const float* x     = (const float*)d_in[0];
    const float* y     = (const float*)d_in[1];
    const int*   force = (const int*)d_in[2];
    const float* eWih0 = (const float*)d_in[3];
    const float* eWhh0 = (const float*)d_in[4];
    const float* ebih0 = (const float*)d_in[5];
    const float* ebhh0 = (const float*)d_in[6];
    const float* eWih  = (const float*)d_in[7];
    const float* eWhh  = (const float*)d_in[8];
    const float* ebih  = (const float*)d_in[9];
    const float* ebhh  = (const float*)d_in[10];
    const float* dWih0 = (const float*)d_in[11];
    const float* dWhh0 = (const float*)d_in[12];
    const float* dbih0 = (const float*)d_in[13];
    const float* dbhh0 = (const float*)d_in[14];
    const float* dWih  = (const float*)d_in[15];
    const float* dWhh  = (const float*)d_in[16];
    const float* dbih  = (const float*)d_in[17];
    const float* dbhh  = (const float*)d_in[18];
    const float* Wu    = (const float*)d_in[19];
    const float* bu    = (const float*)d_in[20];

    char* ws = (char*)d_ws;
    size_t off = 0;
    auto alloc = [&](size_t bytes) {
        size_t o = off; off += (bytes + 255) & ~(size_t)255; return o;
    };
    const size_t szW33 = (size_t)256 * 33 * 512 * 2;   // K=1056 (enc L0)
    const size_t szW64 = (size_t)256 * 64 * 512 * 2;   // K=2048 (fused Wih|Whh)
    const size_t szW32 = (size_t)256 * 32 * 512 * 2;   // K=1024 (dec L0 Whh)
    const size_t szHP  = (size_t)32 * 8 * 32 * 16 * 2; // packed h (256 KB)
    const size_t szC   = (size_t)BDIM * HDIM * 4;

    const size_t o_wE0 = alloc(szW33), o_wE1 = alloc(szW64), o_wE2 = alloc(szW64);
    const size_t o_wD0 = alloc(szW32), o_wD1 = alloc(szW64), o_wD2 = alloc(szW64);
    const size_t o_bE0 = alloc(G4H * 4), o_bE1 = alloc(G4H * 4), o_bE2 = alloc(G4H * 4);
    const size_t o_bD0 = alloc(G4H * 4), o_bD1 = alloc(G4H * 4), o_bD2 = alloc(G4H * 4);
    const size_t o_xp  = alloc((size_t)TDIM * 8 * 32 * 16 * 2);
    const size_t state0 = off;
    size_t o_h[3][2];
    for (int l = 0; l < 3; ++l) for (int pbuf = 0; pbuf < 2; ++pbuf) o_h[l][pbuf] = alloc(szHP);
    const size_t o_c0 = alloc(szC), o_c1 = alloc(szC), o_c2 = alloc(szC);
    const size_t o_hp = alloc(szC);
    const size_t o_xs = alloc(512);
    const size_t o_bar = alloc(256);
    const size_t state_bytes = off - state0;

    hipMemsetAsync(ws + state0, 0, state_bytes, stream);                 // h,c,barriers
    hipMemsetAsync(d_out, 0, (size_t)out_size * sizeof(float), stream);  // t=0 row = 0

    // ---- pack weights/biases/x into WMMA fragment layouts ----
    const size_t WMAT = (size_t)G4H * HDIM;
    {
        int tot = G4H * 1056;
        pack_weights_kernel<<<(tot + 255) / 256, 256, 0, stream>>>(
            eWih0, 32, eWhh0, HDIM, (unsigned short*)(ws + o_wE0), 33, tot);
        tot = G4H * 2048;
        pack_weights_kernel<<<(tot + 255) / 256, 256, 0, stream>>>(
            eWih, HDIM, eWhh, HDIM, (unsigned short*)(ws + o_wE1), 64, tot);
        pack_weights_kernel<<<(tot + 255) / 256, 256, 0, stream>>>(
            eWih + WMAT, HDIM, eWhh + WMAT, HDIM, (unsigned short*)(ws + o_wE2), 64, tot);
        pack_weights_kernel<<<(tot + 255) / 256, 256, 0, stream>>>(
            dWih, HDIM, dWhh, HDIM, (unsigned short*)(ws + o_wD1), 64, tot);
        pack_weights_kernel<<<(tot + 255) / 256, 256, 0, stream>>>(
            dWih + WMAT, HDIM, dWhh + WMAT, HDIM, (unsigned short*)(ws + o_wD2), 64, tot);
        tot = G4H * 1024;
        pack_weights_kernel<<<(tot + 255) / 256, 256, 0, stream>>>(
            nullptr, 0, dWhh0, HDIM, (unsigned short*)(ws + o_wD0), 32, tot);
    }
    pack_bias_kernel<<<G4H / 256, 256, 0, stream>>>(ebih0, ebhh0, (float*)(ws + o_bE0));
    pack_bias_kernel<<<G4H / 256, 256, 0, stream>>>(ebih, ebhh, (float*)(ws + o_bE1));
    pack_bias_kernel<<<G4H / 256, 256, 0, stream>>>(ebih + G4H, ebhh + G4H, (float*)(ws + o_bE2));
    pack_bias_kernel<<<G4H / 256, 256, 0, stream>>>(dbih0, dbhh0, (float*)(ws + o_bD0));
    pack_bias_kernel<<<G4H / 256, 256, 0, stream>>>(dbih, dbhh, (float*)(ws + o_bD1));
    pack_bias_kernel<<<G4H / 256, 256, 0, stream>>>(dbih + G4H, dbhh + G4H, (float*)(ws + o_bD2));
    {
        const int tot = TDIM * 8 * 32 * 16;
        pack_x_kernel<<<(tot + 255) / 256, 256, 0, stream>>>(
            x, (unsigned short*)(ws + o_xp));
    }

    // ---- persistent encoder ----
    EncP ep;
    ep.wB0 = (const unsigned short*)(ws + o_wE0);
    ep.wB1 = (const unsigned short*)(ws + o_wE1);
    ep.wB2 = (const unsigned short*)(ws + o_wE2);
    ep.bias0 = (const float*)(ws + o_bE0);
    ep.bias1 = (const float*)(ws + o_bE1);
    ep.bias2 = (const float*)(ws + o_bE2);
    ep.xp = (const unsigned short*)(ws + o_xp);
    for (int pbuf = 0; pbuf < 2; ++pbuf) {
        ep.h0[pbuf] = (unsigned short*)(ws + o_h[0][pbuf]);
        ep.h1[pbuf] = (unsigned short*)(ws + o_h[1][pbuf]);
        ep.h2[pbuf] = (unsigned short*)(ws + o_h[2][pbuf]);
    }
    ep.c0 = (float*)(ws + o_c0); ep.c1 = (float*)(ws + o_c1); ep.c2 = (float*)(ws + o_c2);
    ep.bar = (unsigned*)(ws + o_bar);
    enc_kernel<<<NWG, NTHREADS, 0, stream>>>(ep);

    // ---- persistent decoder (shares state buffers: enc final == dec init) ----
    DecP dp;
    dp.wB0 = (const unsigned short*)(ws + o_wD0);
    dp.wB1 = (const unsigned short*)(ws + o_wD1);
    dp.wB2 = (const unsigned short*)(ws + o_wD2);
    dp.bias0 = (const float*)(ws + o_bD0);
    dp.bias1 = (const float*)(ws + o_bD1);
    dp.bias2 = (const float*)(ws + o_bD2);
    dp.wih0col = dWih0;
    dp.Wu = Wu; dp.bu = bu; dp.y = y; dp.force = force;
    for (int pbuf = 0; pbuf < 2; ++pbuf) {
        dp.h0[pbuf] = (unsigned short*)(ws + o_h[0][pbuf]);
        dp.h1[pbuf] = (unsigned short*)(ws + o_h[1][pbuf]);
        dp.h2[pbuf] = (unsigned short*)(ws + o_h[2][pbuf]);
    }
    dp.c0 = (float*)(ws + o_c0); dp.c1 = (float*)(ws + o_c1); dp.c2 = (float*)(ws + o_c2);
    dp.h_plain = (float*)(ws + o_hp);
    dp.x_scalar = (float*)(ws + o_xs);
    dp.out = (float*)d_out;
    dp.bar = (unsigned*)(ws + o_bar + 128);   // separate counter from encoder
    dec_kernel<<<NWG, NTHREADS, 0, stream>>>(dp);
}
